// AttentionLayer_7593502179848
// MI455X (gfx1250) — matrix operand order repaired and checked by reading, not compiled, not run on hardware
//
#include <hip/hip_runtime.h>

// ---------------------------------------------------------------------------
// Bahdanau attention, B=1024, N=512, H=C=512, fp32 in/out.
// Strategy: bf16 WMMA (f32 accum) for the 275-GFLOP GEMM; stream sources from
// HBM exactly twice (score pass + context pass) => ~92us HBM floor @23.3TB/s.
// W2 chunks staged into LDS with the Tensor Data Mover (tensor_load_to_lds).
// ---------------------------------------------------------------------------

#define BB 1024
#define NN 512
#define HH 512
#define CC 512

typedef __attribute__((ext_vector_type(16))) __bf16 v16bf;
typedef __attribute__((ext_vector_type(8)))  float  v8f;
typedef __attribute__((ext_vector_type(4)))  unsigned su4;
typedef __attribute__((ext_vector_type(8)))  unsigned su8;

union AFrag { v16bf v; unsigned u[8]; };
union BFrag { v16bf v; uint4 q[2]; };

__device__ inline unsigned short f2bf(float f) {           // round-to-nearest-even
  unsigned u = __float_as_uint(f);
  u += 0x7FFFu + ((u >> 16) & 1u);
  return (unsigned short)(u >> 16);
}
__device__ inline unsigned pk2bf(float lo, float hi) {
  return (unsigned)f2bf(lo) | ((unsigned)f2bf(hi) << 16);
}
__device__ inline float lane_xor_add(float x, int mask, int lane) {
  int p = __builtin_amdgcn_ds_bpermute((lane ^ mask) << 2, __float_as_int(x));
  return x + __int_as_float(p);
}

// ---------------------------------------------------------------------------
// Kernel 0: W2 fp32 [K=512][C=512] -> bf16, pre-swizzled into WMMA B-fragment
// order. Tile T = cchunk*64 + kb*4 + j covers K rows [kb*32,+32), cols
// [(cchunk*4+j)*16,+16). Within a tile: lane l (col = l&15), element e:
// K = kb*32 + (l>>4)*16 + e (contiguous k order, same as the A fragments). Each 64-col chunk
// is a contiguous 64KB block => one TDM descriptor stages it to LDS.
// ---------------------------------------------------------------------------
__global__ __launch_bounds__(256) void swizzle_w2_kernel(
    const float* __restrict__ w2, unsigned* __restrict__ out) {
  int g  = blockIdx.x * 256 + threadIdx.x;   // 0..131071 (uint pairs)
  int e2 = g & 7;
  int l  = (g >> 3) & 31;
  int T  = g >> 8;
  int j  = T & 3;
  int kb = (T >> 2) & 15;
  int cc = T >> 6;
  int c  = (cc * 4 + j) * 16 + (l & 15);
  int k  = kb * 32 + ((l >> 4) ? 16 : 0) + e2 * 2;
  out[g] = pk2bf(w2[k * CC + c], w2[(k + 1) * CC + c]);
}

// ---------------------------------------------------------------------------
// Kernel 1: t = target @ W1 + b1   [1024, 512] fp32 (0.5 GFLOP, VALU).
// ---------------------------------------------------------------------------
__global__ __launch_bounds__(256) void t_kernel(
    const float* __restrict__ target, const float* __restrict__ w1,
    const float* __restrict__ w1b, float* __restrict__ t_ws) {
  __shared__ float st[HH];
  int b = blockIdx.x >> 1;
  int c = (blockIdx.x & 1) * 256 + threadIdx.x;
  st[threadIdx.x]       = target[b * HH + threadIdx.x];
  st[threadIdx.x + 256] = target[b * HH + 256 + threadIdx.x];
  __syncthreads();
  float acc = w1b[c];
#pragma unroll 8
  for (int h = 0; h < HH; ++h) acc += st[h] * w1[h * CC + c];
  t_ws[b * CC + c] = acc;
}

// ---------------------------------------------------------------------------
// Kernel 2: fused score kernel (bf16 WMMA + TDM staging).
// WG = 256 thr = 8 waves, owns 128 consecutive rows m (one batch b).
// Wave w owns rows [m0+16w, m0+16w+16): A tile 16x512 held in registers as
// 16 v16bf fragments, loaded once from HBM (fp32 -> bf16 cvt inline).
// Loop 8 chunks of 64 C-columns: wave 0 issues one tensor_load_to_lds for the
// contiguous 64KB swizzled-W2 chunk, waits TENSORcnt, barrier; then 16
// K-steps x 4 col-tiles of v_wmma_f32_16x16x32_bf16; fused epilogue:
// tanh(t + b2 + s) . V per lane, butterfly-reduced across the 16 col-lanes.
// ---------------------------------------------------------------------------
__global__ __launch_bounds__(256) void score_kernel(
    const float* __restrict__ src, const unsigned* __restrict__ w2s,
    const float* __restrict__ t_ws, const float* __restrict__ w2b,
    const float* __restrict__ vw, const float* __restrict__ vb,
    float* __restrict__ scores) {
  __shared__ uint4 ldsB[4096];                 // 64 KB B chunk
  const int tid  = threadIdx.x;
  const int wave = tid >> 5;
  const int lane = tid & 31;
  const int half = lane >> 4;
  const int lr   = lane & 15;
  const int m0   = blockIdx.x * 128;           // 4 WGs per batch row-block
  const int b    = m0 >> 9;
  const int row  = m0 + wave * 16 + lr;        // this lane's A row (M = lr)
  const float* rowp = src + (size_t)row * HH;

  // ---- load + convert A: 16 fragments (M=16 x K=32 each), kept in VGPRs ----
  AFrag af[16];
#pragma unroll
  for (int kb = 0; kb < 16; ++kb) {
    const int base = kb * 32 + half * 16;      // A layout: K = e + half*16
#pragma unroll
    for (int p = 0; p < 2; ++p) {
      const float4* fp = (const float4*)(rowp + base + p * 8);
      float4 x0 = fp[0], x1 = fp[1];
      af[kb].u[p * 4 + 0] = pk2bf(x0.x, x0.y);
      af[kb].u[p * 4 + 1] = pk2bf(x0.z, x0.w);
      af[kb].u[p * 4 + 2] = pk2bf(x1.x, x1.y);
      af[kb].u[p * 4 + 3] = pk2bf(x1.z, x1.w);
    }
  }

  const unsigned lds0 = (unsigned)(uintptr_t)(void*)&ldsB[0]; // LDS byte addr (low 32b of aperture VA)

  float y[8];
#pragma unroll
  for (int v = 0; v < 8; ++v) y[v] = 0.0f;

  for (int ch = 0; ch < 8; ++ch) {
    __syncthreads();                           // all waves done reading prev chunk
    if (wave == 0) {
      // --- Tensor Data Mover: 64KB contiguous chunk -> LDS (1 instruction) ---
      // D# group0: count=1 | lds_addr | global_addr | type=2
      unsigned long long ga =
          (unsigned long long)(uintptr_t)(w2s + (size_t)ch * 16384);
      su4 g0;
      g0[0] = 1u;                               // count=1, user descriptor
      g0[1] = lds0;                             // lds_addr (bytes)
      g0[2] = (unsigned)ga;                     // global_addr[31:0]
      g0[3] = (unsigned)(ga >> 32) | 0x80000000u; // global_addr[56:32] | type=2
      // D# group1: data_size=8B; tile 8192x1; tensor_dim0=stride0=8192
      su8 g1;
      g1[0] = 0x00030000u;                      // wg_mask=0, data_size=3 (8B)
      g1[1] = 0x20000000u;                      // tensor_dim0[15:0]=8192 -> [31:16]
      g1[2] = 0x00010000u;                      // tensor_dim0[31:16]=0, tensor_dim1=1
      g1[3] = 0x20000000u;                      // tile_dim0=8192 -> [31:16]
      g1[4] = 0x00000001u;                      // tile_dim1=1, tile_dim2=0
      g1[5] = 8192u;                            // tensor_dim0_stride[31:0]
      g1[6] = 0u;                               // stride0[47:32]=0, stride1 lo=0
      g1[7] = 0u;
      asm volatile("tensor_load_to_lds %0, %1" :: "s"(g0), "s"(g1) : "memory");
      __builtin_amdgcn_s_wait_tensorcnt(0);     // TDM complete (issuing wave)
    }
    __syncthreads();                           // chunk visible to all waves

    v8f acc[4];
#pragma unroll
    for (int j = 0; j < 4; ++j)
#pragma unroll
      for (int e = 0; e < 8; ++e) acc[j][e] = 0.0f;

#pragma unroll
    for (int kb = 0; kb < 16; ++kb) {
      v16bf a = af[kb].v;
#pragma unroll
      for (int j = 0; j < 4; ++j) {
        BFrag bf;
        int bi  = (kb * 4 + j) * 64 + lane * 2;
        bf.q[0] = ldsB[bi];
        bf.q[1] = ldsB[bi + 1];
        acc[j] = __builtin_amdgcn_wmma_f32_16x16x32_bf16(
            false, a, false, bf.v, (short)0, acc[j], false, false);
      }
    }

    // fused epilogue: y += V[c] * tanh(t[b,c] + b2[c] + s)
#pragma unroll
    for (int j = 0; j < 4; ++j) {
      int col  = ch * 64 + j * 16 + lr;        // C/D layout: col = lane%16
      float tv = t_ws[b * CC + col] + w2b[col];
      float vv = vw[col];
#pragma unroll
      for (int v = 0; v < 8; ++v)              // row = v + 8*half
        y[v] += vv * tanhf(acc[j][v] + tv);
    }
  }

  // reduce the 16 column-lanes within each half-wave (masks < 16 stay in half)
#pragma unroll
  for (int v = 0; v < 8; ++v) {
    float t = y[v];
    t = lane_xor_add(t, 1, lane);
    t = lane_xor_add(t, 2, lane);
    t = lane_xor_add(t, 4, lane);
    t = lane_xor_add(t, 8, lane);
    y[v] = t;
  }
  if (lr == 0) {                               // lanes 0 & 16 hold rows 0-7 / 8-15
    int n0     = (m0 & (NN - 1)) + wave * 16 + half * 8;
    float bias = vb[0];
#pragma unroll
    for (int v = 0; v < 8; ++v)
      scores[b * NN + n0 + v] = (y[v] + bias) * 0.04419417382415922f; // 1/sqrt(512)
  }
}

// ---------------------------------------------------------------------------
// Kernel 3: softmax over N=512 per batch row.
// ---------------------------------------------------------------------------
__global__ __launch_bounds__(256) void softmax_kernel(
    const float* __restrict__ scores, float* __restrict__ attn) {
  __shared__ float red[256];
  int b = blockIdx.x, tid = threadIdx.x;
  float x0 = scores[b * NN + tid], x1 = scores[b * NN + 256 + tid];
  red[tid] = fmaxf(x0, x1);
  __syncthreads();
  for (int s = 128; s > 0; s >>= 1) {
    if (tid < s) red[tid] = fmaxf(red[tid], red[tid + s]);
    __syncthreads();
  }
  float m = red[0];
  __syncthreads();
  float e0 = __expf(x0 - m), e1 = __expf(x1 - m);
  red[tid] = e0 + e1;
  __syncthreads();
  for (int s = 128; s > 0; s >>= 1) {
    if (tid < s) red[tid] += red[tid + s];
    __syncthreads();
  }
  float inv = 1.0f / red[0];
  attn[b * NN + tid]       = e0 * inv;
  attn[b * NN + 256 + tid] = e1 * inv;
}

// ---------------------------------------------------------------------------
// Kernel 4: context[b,h] = sum_n attn[b,n] * sources[b,n,h]  (2nd full stream
// of sources, fp32 FMA, coalesced float2; HBM-bound by design).
// ---------------------------------------------------------------------------
__global__ __launch_bounds__(256) void context_kernel(
    const float* __restrict__ src, const float* __restrict__ attn,
    float* __restrict__ ctx) {
  __shared__ float w[NN];
  int b = blockIdx.x, tid = threadIdx.x;
  w[tid]       = attn[b * NN + tid];
  w[tid + 256] = attn[b * NN + 256 + tid];
  __syncthreads();
  const float2* p = (const float2*)(src + (size_t)b * NN * HH) + tid;
  float2 acc = {0.0f, 0.0f};
#pragma unroll 4
  for (int n = 0; n < NN; ++n) {
    float2 sv = p[(size_t)n * 256];
    float wn  = w[n];
    acc.x += wn * sv.x;
    acc.y += wn * sv.y;
  }
  ((float2*)(ctx + b * HH))[tid] = acc;
}

// ---------------------------------------------------------------------------
extern "C" void kernel_launch(void* const* d_in, const int* in_sizes, int n_in,
                              void* d_out, int out_size, void* d_ws, size_t ws_size,
                              hipStream_t stream) {
  const float* target = (const float*)d_in[0];
  const float* sources = (const float*)d_in[1];
  const float* W1w = (const float*)d_in[2];
  const float* W1b = (const float*)d_in[3];
  const float* W2w = (const float*)d_in[4];
  const float* W2b = (const float*)d_in[5];
  const float* Vw  = (const float*)d_in[6];
  const float* Vb  = (const float*)d_in[7];

  float* ctx  = (float*)d_out;                  // [B*H] context_vector
  float* attn = (float*)d_out + BB * HH;        // [B*N] attention_weights

  char* ws = (char*)d_ws;                       // 4.5 MiB used
  float*    t_ws   = (float*)ws;                               // 2 MiB
  float*    scores = (float*)(ws + 2u * 1024 * 1024);          // 2 MiB
  unsigned* w2s    = (unsigned*)(ws + 4u * 1024 * 1024);       // 512 KiB

  swizzle_w2_kernel<<<dim3(512),  dim3(256), 0, stream>>>(W2w, w2s);
  t_kernel        <<<dim3(2048), dim3(256), 0, stream>>>(target, W1w, W1b, t_ws);
  score_kernel    <<<dim3(4096), dim3(256), 0, stream>>>(sources, w2s, t_ws, W2b, Vw, Vb, scores);
  softmax_kernel  <<<dim3(1024), dim3(256), 0, stream>>>(scores, attn);
  context_kernel  <<<dim3(1024), dim3(256), 0, stream>>>(sources, attn, ctx);
}
